// CandidateEmbedder_35845797053089
// MI455X (gfx1250) — compile-verified, matrix operand order checked by reading, not run on hardware
//
#include <hip/hip_runtime.h>
#include <hip/hip_bf16.h>
#include <math.h>

// ---------------------------------------------------------------------------
// BigBird encoder on MI455X (gfx1250): bf16 WMMA GEMMs + flash block-sparse
// attention. wave32 everywhere; WMMA shape 16x16x32 bf16 -> f32 acc.
// New this round: async bias staging via global_load_async_to_lds_b32 +
// s_wait_asynccnt (ASYNCcnt path), latency-hidden across the whole K loop.
// ---------------------------------------------------------------------------

typedef __attribute__((ext_vector_type(16))) __bf16 v16bf;
typedef __attribute__((ext_vector_type(8)))  __bf16 v8bf;
typedef __attribute__((ext_vector_type(8)))  float  v8f;

#define S_LEN 4096
#define D_MOD 768
#define N_HEAD 12
#define H_DIM 64
#define N_BLK 64
#define BLK_S 64
#define D_FF 3072
#define LAYERS 2
#define LN_EPS 1e-12f

static __device__ __forceinline__ v16bf mk16(v8bf lo, v8bf hi) {
    v16bf r;
#pragma unroll
    for (int i = 0; i < 8; ++i) { r[i] = lo[i]; r[i + 8] = hi[i]; }
    return r;
}

static __device__ __forceinline__ v8f wmma_bf16(v16bf a, v16bf b, v8f c) {
    // D = A(16x32 bf16) x B(32x16 bf16) + C(16x16 f32)
    return __builtin_amdgcn_wmma_f32_16x16x32_bf16(false, a, false, b,
                                                   (short)0, c, false, false);
}

static __device__ __forceinline__ v8f zero8() {
    v8f z = {0.f, 0.f, 0.f, 0.f, 0.f, 0.f, 0.f, 0.f};
    return z;
}

static __device__ __forceinline__ float block_reduce_sum(float v, float* sh) {
    int tid = threadIdx.x;
    sh[tid] = v;
    __syncthreads();
    for (int off = 128; off > 0; off >>= 1) {
        if (tid < off && tid + off < 256) sh[tid] += sh[tid + off];
        __syncthreads();
    }
    float r = sh[0];
    __syncthreads();
    return r;
}

// ---------------------------------------------------------------------------
// 1) Embedding lookup + LayerNorm. One block per token, 256 threads.
// ---------------------------------------------------------------------------
__global__ void k_embed(const int* __restrict__ ids,
                        const float* __restrict__ ew,
                        const float* __restrict__ ep,
                        const float* __restrict__ gs,
                        const float* __restrict__ gb,
                        float* __restrict__ h, __bf16* __restrict__ hbf) {
    __shared__ float red[256];
    int srow = blockIdx.x;
    int tid = threadIdx.x;
    int id = ids[srow];
    float v[3];
    float s = 0.f;
#pragma unroll
    for (int i = 0; i < 3; ++i) {
        int d = tid + i * 256;
        v[i] = ew[(size_t)id * D_MOD + d] + ep[(size_t)srow * D_MOD + d];
        s += v[i];
    }
    float mean = block_reduce_sum(s, red) * (1.f / D_MOD);
    float q = 0.f;
#pragma unroll
    for (int i = 0; i < 3; ++i) { float d = v[i] - mean; q += d * d; }
    float var = block_reduce_sum(q, red) * (1.f / D_MOD);
    float inv = rsqrtf(var + LN_EPS);
#pragma unroll
    for (int i = 0; i < 3; ++i) {
        int d = tid + i * 256;
        float y = (v[i] - mean) * inv * gs[d] + gb[d];
        h[(size_t)srow * D_MOD + d] = y;
        hbf[(size_t)srow * D_MOD + d] = (__bf16)y;
    }
}

// ---------------------------------------------------------------------------
// 2) y = LN(x + res); writes f32 + bf16.
// ---------------------------------------------------------------------------
__global__ void k_ln_res(const float* __restrict__ x,
                         const float* __restrict__ res,
                         const float* __restrict__ gs,
                         const float* __restrict__ gb,
                         float* __restrict__ h, __bf16* __restrict__ hbf) {
    __shared__ float red[256];
    int srow = blockIdx.x;
    int tid = threadIdx.x;
    float v[3];
    float s = 0.f;
#pragma unroll
    for (int i = 0; i < 3; ++i) {
        int d = tid + i * 256;
        v[i] = x[(size_t)srow * D_MOD + d] + res[(size_t)srow * D_MOD + d];
        s += v[i];
    }
    float mean = block_reduce_sum(s, red) * (1.f / D_MOD);
    float q = 0.f;
#pragma unroll
    for (int i = 0; i < 3; ++i) { float d = v[i] - mean; q += d * d; }
    float var = block_reduce_sum(q, red) * (1.f / D_MOD);
    float inv = rsqrtf(var + LN_EPS);
#pragma unroll
    for (int i = 0; i < 3; ++i) {
        int d = tid + i * 256;
        float y = (v[i] - mean) * inv * gs[d] + gb[d];
        h[(size_t)srow * D_MOD + d] = y;
        hbf[(size_t)srow * D_MOD + d] = (__bf16)y;
    }
}

// ---------------------------------------------------------------------------
// 3) Weight convert + transpose: f32 [K,N] -> bf16 [N,K].
// ---------------------------------------------------------------------------
__global__ void k_wT(const float* __restrict__ w, __bf16* __restrict__ wt,
                     int K, int N) {
    size_t e = (size_t)blockIdx.x * 256 + threadIdx.x;
    if (e < (size_t)K * N) {
        int k = (int)(e / N);
        int n = (int)(e % N);
        wt[(size_t)n * K + k] = (__bf16)w[e];
    }
}

// ---------------------------------------------------------------------------
// 4) WMMA GEMM: C[M,N] = A_bf16[M,K] * Wt_bf16[N,K] + bias, optional GELU.
//    Block tile 64x128 (8 waves, 32x32 each). Outputs: f32, bf16, bf16^T.
//    Bias slice staged LDS-side with an async load issued in the prologue;
//    the ASYNCcnt wait lands only at the epilogue (latency hidden by K loop).
// ---------------------------------------------------------------------------
__global__ void k_gemm(const __bf16* __restrict__ A,
                       const __bf16* __restrict__ W,
                       const float* __restrict__ bias,
                       float* __restrict__ outF, __bf16* __restrict__ outB,
                       __bf16* __restrict__ outBT,
                       int M, int N, int K, int act) {
    __shared__ alignas(16) float sbias[128];
    int tid = threadIdx.x;
    int wv = tid >> 5;
    int lane = tid & 31;
    int half = lane >> 4;
    int ln = lane & 15;
    int m0 = blockIdx.y * 64 + (wv & 1) * 32;
    int nblk = blockIdx.x * 128;
    int n0 = nblk + (wv >> 1) * 32;

    // --- async bias stage: 128 lanes each pull one f32 into LDS (ASYNCcnt)
    if (tid < 128) {
        unsigned int ldsa = (unsigned int)(size_t)(&sbias[tid]);
        unsigned long long ga = (unsigned long long)(size_t)(bias + nblk + tid);
        asm volatile("global_load_async_to_lds_b32 %0, %1, off"
                     :: "v"(ldsa), "v"(ga) : "memory");
    }

    v8f acc[2][2];
#pragma unroll
    for (int i = 0; i < 2; ++i)
#pragma unroll
        for (int j = 0; j < 2; ++j) acc[i][j] = zero8();

    for (int k0 = 0; k0 < K; k0 += 32) {
        v16bf a[2], b[2];
#pragma unroll
        for (int ti = 0; ti < 2; ++ti) {
            const __bf16* ap = A + (size_t)(m0 + ti * 16 + ln) * K + k0 + half * 8;
            a[ti] = mk16(*reinterpret_cast<const v8bf*>(ap),
                         *reinterpret_cast<const v8bf*>(ap + 16));
        }
#pragma unroll
        for (int tj = 0; tj < 2; ++tj) {
            const __bf16* wp = W + (size_t)(n0 + tj * 16 + ln) * K + k0 + half * 8;
            b[tj] = mk16(*reinterpret_cast<const v8bf*>(wp),
                         *reinterpret_cast<const v8bf*>(wp + 16));
        }
        if (k0 + 32 < K) {
            __builtin_prefetch(A + (size_t)(m0 + ln) * K + k0 + 32, 0, 0);
            __builtin_prefetch(W + (size_t)(n0 + ln) * K + k0 + 32, 0, 0);
        }
#pragma unroll
        for (int ti = 0; ti < 2; ++ti)
#pragma unroll
            for (int tj = 0; tj < 2; ++tj)
                acc[ti][tj] = wmma_bf16(a[ti], b[tj], acc[ti][tj]);
    }

    // --- drain the async bias copy, make visible block-wide
    asm volatile("s_wait_asynccnt 0x0" ::: "memory");
    __syncthreads();

#pragma unroll
    for (int ti = 0; ti < 2; ++ti)
#pragma unroll
        for (int tj = 0; tj < 2; ++tj)
#pragma unroll
            for (int r = 0; r < 8; ++r) {
                int row = m0 + ti * 16 + r + half * 8;
                int col = n0 + tj * 16 + ln;
                float v = acc[ti][tj][r] + sbias[col - nblk];
                if (act == 1)  // exact GELU
                    v = 0.5f * v * (1.f + erff(v * 0.70710678118654752f));
                if (outF) outF[(size_t)row * N + col] = v;
                if (outB) outB[(size_t)row * N + col] = (__bf16)v;
                if (outBT) outBT[(size_t)col * M + row] = (__bf16)v;
            }
}

// ---------------------------------------------------------------------------
// 5) BigBird index table: [62][8] = {g0, gN, w-1, w, w+1, r0, r1, r2}
// ---------------------------------------------------------------------------
__global__ void k_build_idx(const int* __restrict__ rb, int* __restrict__ idx) {
    int r = threadIdx.x;
    if (r < N_BLK - 2) {
        idx[r * 8 + 0] = 0;
        idx[r * 8 + 1] = N_BLK - 1;
        idx[r * 8 + 2] = r;
        idx[r * 8 + 3] = r + 1;
        idx[r * 8 + 4] = r + 2;
        idx[r * 8 + 5] = rb[r * 3 + 0];
        idx[r * 8 + 6] = rb[r * 3 + 1];
        idx[r * 8 + 7] = rb[r * 3 + 2];
    }
}

// ---------------------------------------------------------------------------
// 6) Flash block attention. grid=(nQ, H), block=128 (4 waves).
//    mode 0: middle blocks, qb = blockIdx.x+1, key blocks from idxTab (Kb=8).
//    mode 1: global rows,  qb in {0, NB-1}, key blocks 0..63 (Kb=64).
//    Q/K from bf16 [S,D]; V transposed bf16 [D,S]; output bf16 [S,D].
// ---------------------------------------------------------------------------
__global__ void k_attn(const __bf16* __restrict__ qbf,
                       const __bf16* __restrict__ kbf,
                       const __bf16* __restrict__ vtb,
                       const float* __restrict__ mask,
                       const int* __restrict__ idxTab, int Kb, int mode,
                       __bf16* __restrict__ outbf, float scale) {
    __shared__ alignas(16) __bf16 plds[4 * 16 * 64];  // per-wave P staging
    int head = blockIdx.y;
    int qi = blockIdx.x;
    int qb = (mode == 0) ? (qi + 1) : (qi == 0 ? 0 : N_BLK - 1);
    int wv = threadIdx.x >> 5;
    int lane = threadIdx.x & 31;
    int half = lane >> 4;
    int ln = lane & 15;

    // Q A-fragments for this wave's 16 query rows (loaded once).
    int qrow = qb * BLK_S + wv * 16 + ln;
    const __bf16* qp = qbf + (size_t)qrow * D_MOD + head * H_DIM;
    v16bf qa[2];
#pragma unroll
    for (int c = 0; c < 2; ++c) {
        const __bf16* p = qp + c * 32 + half * 8;
        qa[c] = mk16(*reinterpret_cast<const v8bf*>(p),
                     *reinterpret_cast<const v8bf*>(p + 16));
    }

    v8f o[4];
    float rm[8], rl[8];
#pragma unroll
    for (int t = 0; t < 4; ++t) o[t] = zero8();
#pragma unroll
    for (int r = 0; r < 8; ++r) { rm[r] = -1e30f; rl[r] = 0.f; }

    for (int it = 0; it < Kb; ++it) {
        int kb = idxTab ? idxTab[qi * 8 + it] : it;

        // ---- scores S = (Q K^T) * scale + mask
        v8f st[4];
#pragma unroll
        for (int t = 0; t < 4; ++t) {
            int krow = kb * BLK_S + t * 16 + ln;
            const __bf16* kp = kbf + (size_t)krow * D_MOD + head * H_DIM;
            v8f s = zero8();
#pragma unroll
            for (int c = 0; c < 2; ++c) {
                const __bf16* p = kp + c * 32 + half * 8;
                v16bf b = mk16(*reinterpret_cast<const v8bf*>(p),
                               *reinterpret_cast<const v8bf*>(p + 16));
                s = wmma_bf16(qa[c], b, s);
            }
            float madd = (1.f - mask[kb * BLK_S + t * 16 + ln]) * (-1e9f);
#pragma unroll
            for (int r = 0; r < 8; ++r) st[t][r] = s[r] * scale + madd;
        }

        // ---- online softmax row stats (rows r + half*8; 16-lane reductions)
        float nm[8], al[8];
#pragma unroll
        for (int r = 0; r < 8; ++r) {
            float mx = st[0][r];
#pragma unroll
            for (int t = 1; t < 4; ++t) mx = fmaxf(mx, st[t][r]);
#pragma unroll
            for (int off = 1; off < 16; off <<= 1)
                mx = fmaxf(mx, __shfl_xor(mx, off, 32));
            nm[r] = fmaxf(rm[r], mx);
            al[r] = __expf(rm[r] - nm[r]);
            rm[r] = nm[r];
        }
#pragma unroll
        for (int t = 0; t < 4; ++t)
#pragma unroll
            for (int r = 0; r < 8; ++r) st[t][r] = __expf(st[t][r] - nm[r]);
#pragma unroll
        for (int r = 0; r < 8; ++r) {
            float s = st[0][r] + st[1][r] + st[2][r] + st[3][r];
#pragma unroll
            for (int off = 1; off < 16; off <<= 1) s += __shfl_xor(s, off, 32);
            rl[r] = rl[r] * al[r] + s;
        }

        // ---- P (D-layout) -> LDS -> A-layout; rescale O
        __bf16* lp = plds + wv * 1024;
#pragma unroll
        for (int t = 0; t < 4; ++t)
#pragma unroll
            for (int r = 0; r < 8; ++r) {
                int rr = r + half * 8;
                lp[rr * 64 + t * 16 + ln] = (__bf16)st[t][r];
            }
#pragma unroll
        for (int t = 0; t < 4; ++t)
#pragma unroll
            for (int r = 0; r < 8; ++r) o[t][r] *= al[r];

        // ---- O += P @ V  (V^T rows are contiguous key tokens)
#pragma unroll
        for (int c = 0; c < 2; ++c) {
            const __bf16* pp = lp + ln * 64 + c * 32 + half * 8;
            v16bf pa = mk16(*reinterpret_cast<const v8bf*>(pp),
                            *reinterpret_cast<const v8bf*>(pp + 16));
#pragma unroll
            for (int t = 0; t < 4; ++t) {
                int hd = t * 16 + ln;
                const __bf16* vp = vtb + (size_t)(head * H_DIM + hd) * S_LEN +
                                   kb * BLK_S + c * 32 + half * 8;
                v16bf vb = mk16(*reinterpret_cast<const v8bf*>(vp),
                                *reinterpret_cast<const v8bf*>(vp + 16));
                o[t] = wmma_bf16(pa, vb, o[t]);
            }
        }
    }

    // ---- epilogue: O / l -> bf16 [S, D]
#pragma unroll
    for (int t = 0; t < 4; ++t)
#pragma unroll
        for (int r = 0; r < 8; ++r) {
            int rr = r + half * 8;
            int row = qb * BLK_S + wv * 16 + rr;
            int col = head * H_DIM + t * 16 + ln;
            outbf[(size_t)row * D_MOD + col] = (__bf16)(o[t][r] / rl[r]);
        }
}

// ---------------------------------------------------------------------------
// 7) Masked mean pooling over S.
// ---------------------------------------------------------------------------
__global__ void k_pool(const float* __restrict__ h,
                       const float* __restrict__ mask,
                       float* __restrict__ pooled) {
    int j = blockIdx.x * 256 + threadIdx.x;
    if (j < D_MOD) {
        float s = 0.f, ms = 0.f;
        for (int t = 0; t < S_LEN; ++t) {
            float mk = mask[t];
            s += h[(size_t)t * D_MOD + j] * mk;
            ms += mk;
        }
        pooled[j] = s / fmaxf(ms, 1e-9f);
    }
}

// ---------------------------------------------------------------------------
// 8) MLP head: concat(pooled, num) -> mish -> mish -> linear. One block.
// ---------------------------------------------------------------------------
static __device__ __forceinline__ float mish(float x) {
    float sp = log1pf(__expf(x));
    return x * tanhf(sp);
}

__global__ void k_head(const float* __restrict__ pooled,
                       const float* __restrict__ xnum,
                       const float* __restrict__ Wnum, const float* __restrict__ bnum,
                       const float* __restrict__ Wm1, const float* __restrict__ bm1,
                       const float* __restrict__ Wm2, const float* __restrict__ bm2,
                       const float* __restrict__ Wm3, const float* __restrict__ bm3,
                       float* __restrict__ out) {
    __shared__ float z[800];
    __shared__ float z1[400];
    __shared__ float z2[200];
    int tid = threadIdx.x;
    for (int j = tid; j < 800; j += 256) {
        if (j < 768) {
            z[j] = pooled[j];
        } else {
            int c = j - 768;
            float a = bnum[c];
            for (int i = 0; i < 11; ++i) a += xnum[i] * Wnum[i * 32 + c];
            z[j] = a;
        }
    }
    __syncthreads();
    for (int j = tid; j < 400; j += 256) {
        float a = bm1[j];
        for (int i = 0; i < 800; ++i) a += z[i] * Wm1[(size_t)i * 400 + j];
        z1[j] = mish(a);
    }
    __syncthreads();
    for (int j = tid; j < 200; j += 256) {
        float a = bm2[j];
        for (int i = 0; i < 400; ++i) a += z1[i] * Wm2[(size_t)i * 200 + j];
        z2[j] = mish(a);
    }
    __syncthreads();
    if (tid < 100) {
        float a = bm3[tid];
        for (int i = 0; i < 200; ++i) a += z2[i] * Wm3[(size_t)i * 100 + tid];
        out[tid] = a;
    }
}

// ---------------------------------------------------------------------------
// Launch
// ---------------------------------------------------------------------------
extern "C" void kernel_launch(void* const* d_in, const int* in_sizes, int n_in,
                              void* d_out, int out_size, void* d_ws, size_t ws_size,
                              hipStream_t stream) {
    const float* xnum   = (const float*)d_in[0];
    const int*   ids    = (const int*)d_in[1];
    const float* mask   = (const float*)d_in[2];
    const int*   rblk   = (const int*)d_in[3];
    const float* ew     = (const float*)d_in[4];
    const float* ep     = (const float*)d_in[5];
    const float* ln_es  = (const float*)d_in[6];
    const float* ln_eb  = (const float*)d_in[7];
    const float* Wq     = (const float*)d_in[8];
    const float* bq     = (const float*)d_in[9];
    const float* Wk     = (const float*)d_in[10];
    const float* bk     = (const float*)d_in[11];
    const float* Wv     = (const float*)d_in[12];
    const float* bv     = (const float*)d_in[13];
    const float* Wo     = (const float*)d_in[14];
    const float* bo     = (const float*)d_in[15];
    const float* ln1s   = (const float*)d_in[16];
    const float* ln1b   = (const float*)d_in[17];
    const float* W1     = (const float*)d_in[18];
    const float* b1     = (const float*)d_in[19];
    const float* W2     = (const float*)d_in[20];
    const float* b2     = (const float*)d_in[21];
    const float* ln2s   = (const float*)d_in[22];
    const float* ln2b   = (const float*)d_in[23];
    const float* Wnum   = (const float*)d_in[24];
    const float* bnum   = (const float*)d_in[25];
    const float* Wm1    = (const float*)d_in[26];
    const float* bm1    = (const float*)d_in[27];
    const float* Wm2    = (const float*)d_in[28];
    const float* bm2    = (const float*)d_in[29];
    const float* Wm3    = (const float*)d_in[30];
    const float* bm3    = (const float*)d_in[31];

    const size_t SD = (size_t)S_LEN * D_MOD;
    const size_t SF = (size_t)S_LEN * D_FF;

    char* p = (char*)d_ws;
    auto take = [&](size_t bytes) {
        void* r = (void*)p;
        p += (bytes + 255) & ~(size_t)255;
        return r;
    };
    float*  hF   = (float*)take(SD * 4);
    __bf16* hB   = (__bf16*)take(SD * 2);
    __bf16* qB   = (__bf16*)take(SD * 2);
    __bf16* kB   = (__bf16*)take(SD * 2);
    __bf16* vT   = (__bf16*)take(SD * 2);   // [D, S]
    __bf16* aB   = (__bf16*)take(SD * 2);   // attention output bf16
    float*  gF   = (float*)take(SF * 4);    // gemm f32 scratch (max S x DFF)
    __bf16* gB   = (__bf16*)take(SF * 2);   // ffn intermediate bf16
    __bf16* wT   = (__bf16*)take((size_t)D_MOD * D_FF * 2);
    int*    idx  = (int*)take((size_t)(N_BLK - 2) * 8 * 4);
    float*  pool = (float*)take(D_MOD * 4);

    const float scale = 0.125f;  // 1/sqrt(64)

    k_embed<<<S_LEN, 256, 0, stream>>>(ids, ew, ep, ln_es, ln_eb, hF, hB);
    k_build_idx<<<1, 64, 0, stream>>>(rblk, idx);

    for (int l = 0; l < LAYERS; ++l) {
        const float* Wq_l = Wq + (size_t)l * D_MOD * D_MOD;
        const float* Wk_l = Wk + (size_t)l * D_MOD * D_MOD;
        const float* Wv_l = Wv + (size_t)l * D_MOD * D_MOD;
        const float* Wo_l = Wo + (size_t)l * D_MOD * D_MOD;
        const float* W1_l = W1 + (size_t)l * D_MOD * D_FF;
        const float* W2_l = W2 + (size_t)l * D_FF * D_MOD;
        const float* bq_l = bq + (size_t)l * D_MOD;
        const float* bk_l = bk + (size_t)l * D_MOD;
        const float* bv_l = bv + (size_t)l * D_MOD;
        const float* bo_l = bo + (size_t)l * D_MOD;
        const float* b1_l = b1 + (size_t)l * D_FF;
        const float* b2_l = b2 + (size_t)l * D_MOD;

        dim3 gDD(D_MOD / 128, S_LEN / 64);   // N=768
        dim3 gDF(D_FF / 128, S_LEN / 64);    // N=3072
        int wDD = (D_MOD * D_MOD + 255) / 256;
        int wDF = (D_MOD * D_FF + 255) / 256;

        // Q, K, V projections (bf16 outputs; V transposed to [D,S])
        k_wT<<<wDD, 256, 0, stream>>>(Wq_l, wT, D_MOD, D_MOD);
        k_gemm<<<gDD, 256, 0, stream>>>(hB, wT, bq_l, nullptr, qB, nullptr,
                                        S_LEN, D_MOD, D_MOD, 0);
        k_wT<<<wDD, 256, 0, stream>>>(Wk_l, wT, D_MOD, D_MOD);
        k_gemm<<<gDD, 256, 0, stream>>>(hB, wT, bk_l, nullptr, kB, nullptr,
                                        S_LEN, D_MOD, D_MOD, 0);
        k_wT<<<wDD, 256, 0, stream>>>(Wv_l, wT, D_MOD, D_MOD);
        k_gemm<<<gDD, 256, 0, stream>>>(hB, wT, bv_l, nullptr, nullptr, vT,
                                        S_LEN, D_MOD, D_MOD, 0);

        // block-sparse attention: middle blocks + global rows
        k_attn<<<dim3(N_BLK - 2, N_HEAD), 128, 0, stream>>>(
            qB, kB, vT, mask, idx, 8, 0, aB, scale);
        k_attn<<<dim3(2, N_HEAD), 128, 0, stream>>>(
            qB, kB, vT, mask, nullptr, N_BLK, 1, aB, scale);

        // output projection + LN residual
        k_wT<<<wDD, 256, 0, stream>>>(Wo_l, wT, D_MOD, D_MOD);
        k_gemm<<<gDD, 256, 0, stream>>>(aB, wT, bo_l, gF, nullptr, nullptr,
                                        S_LEN, D_MOD, D_MOD, 0);
        k_ln_res<<<S_LEN, 256, 0, stream>>>(gF, hF, ln1s + (size_t)l * D_MOD,
                                            ln1b + (size_t)l * D_MOD, hF, hB);

        // FFN: GELU fused in first GEMM
        k_wT<<<wDF, 256, 0, stream>>>(W1_l, wT, D_MOD, D_FF);
        k_gemm<<<gDF, 256, 0, stream>>>(hB, wT, b1_l, nullptr, gB, nullptr,
                                        S_LEN, D_FF, D_MOD, 1);
        k_wT<<<wDF, 256, 0, stream>>>(W2_l, wT, D_FF, D_MOD);
        k_gemm<<<gDD, 256, 0, stream>>>(gB, wT, b2_l, gF, nullptr, nullptr,
                                        S_LEN, D_MOD, D_FF, 0);
        k_ln_res<<<S_LEN, 256, 0, stream>>>(gF, hF, ln2s + (size_t)l * D_MOD,
                                            ln2b + (size_t)l * D_MOD, hF, hB);
    }

    k_pool<<<3, 256, 0, stream>>>(hF, mask, pool);
    k_head<<<1, 256, 0, stream>>>(pool, xnum, Wnum, bnum, Wm1, bm1, Wm2, bm2,
                                  Wm3, bm3, (float*)d_out);
}
